// SaccadicLayer_48773648613559
// MI455X (gfx1250) — compile-verified, matrix operand order checked by reading, not run on hardware
//
#include <hip/hip_runtime.h>
#include <math.h>

typedef __bf16 bf16;
typedef __attribute__((ext_vector_type(16))) __bf16 v16bf;
typedef __attribute__((ext_vector_type(8)))  __bf16 v8bf;
typedef __attribute__((ext_vector_type(8)))  float  v8f;

static __device__ __forceinline__ float gelu_f(float x) {
    return 0.5f * x * (1.0f + erff(x * 0.70710678118654752440f));
}

// ------------------------------------------------------------------
// generic f32 -> bf16 conversion
// ------------------------------------------------------------------
__global__ void k_cvt_bf16(const float* __restrict__ s, bf16* __restrict__ d, long n)
{
    long i = (long)blockIdx.x * blockDim.x + threadIdx.x;
    if (i < n) d[i] = (bf16)s[i];
}

// conv_w [256][256][64] (o,c,k) -> bf16 [o][k*256+c] so the block-conv is a
// contiguous-K GEMM against xm rows (64 tokens x 256 ch = 16384 contiguous).
__global__ void k_permute_convw(const float* __restrict__ src, bf16* __restrict__ dst)
{
    long i = (long)blockIdx.x * blockDim.x + threadIdx.x;  // = o*16384 + k*256 + c
    if (i >= 256L * 16384L) return;
    int o   = (int)(i >> 14);
    int rem = (int)(i & 16383);
    int k   = rem >> 8;
    int c   = rem & 255;
    dst[i] = (bf16)src[((long)o * 256 + c) * 64 + k];
}

// ------------------------------------------------------------------
// WMMA bf16 GEMM:  C[M,N] = A[M,K] * W[N,K]^T + bias
// one wave -> one 32x64 block (2 M-tiles x 4 N-tiles, 8 accumulators),
// K-step 32, B-fragments shared between the two M-tiles.
// MODE: 0 = +bias store f32 ; 1 = +bias store bf16 ;
//       2 = +bias gelu store bf16 ; 3 = +bias +res store f32
// AF32: A is fp32 in memory, converted to bf16 in-register.
// Requires: M % 32 == 0, N % 64 == 0, K % 32 == 0.
// ------------------------------------------------------------------
template<int MODE, bool AF32>
__global__ void k_gemm(const void* __restrict__ Ap, long lda,
                       const bf16* __restrict__ W, long ldw,
                       const float* __restrict__ bias,
                       void* __restrict__ Cp, long ldc,
                       const float* __restrict__ res,
                       int M, int N, int K)
{
    int wid = blockIdx.x * (blockDim.x >> 5) + (threadIdx.x >> 5);
    int nstrips = N >> 6;
    int total   = (M >> 5) * nstrips;
    if (wid >= total) return;
    int mt = wid / nstrips;
    int ns = wid - mt * nstrips;
    int m0 = mt << 5, n0 = ns << 6;
    int lane = threadIdx.x & 31;
    int g = lane >> 4;        // lane-half selects K subset
    int r = lane & 15;        // A: row within tile ; B: column within 16-wide tile

    v8f acc[2][4];
#pragma unroll
    for (int ti = 0; ti < 2; ++ti)
#pragma unroll
        for (int j = 0; j < 4; ++j)
#pragma unroll
            for (int e = 0; e < 8; ++e) acc[ti][j][e] = 0.0f;

    long aoff0 = (long)(m0 + r) * lda;
    long aoff1 = (long)(m0 + 16 + r) * lda;
    for (int k = 0; k < K; k += 32) {
        int kb = k + (g << 3);   // lanes 0-15: K {k..k+7, k+16..k+23}; 16-31: +8
        v16bf a0, a1;
        if constexpr (AF32) {
            const float* A = (const float*)Ap;
            v8f f0 = *(const v8f*)(A + aoff0 + kb);
            v8f f1 = *(const v8f*)(A + aoff0 + kb + 16);
            v8f f2 = *(const v8f*)(A + aoff1 + kb);
            v8f f3 = *(const v8f*)(A + aoff1 + kb + 16);
#pragma unroll
            for (int e = 0; e < 8; ++e) {
                a0[e] = (bf16)f0[e]; a0[e + 8] = (bf16)f1[e];
                a1[e] = (bf16)f2[e]; a1[e + 8] = (bf16)f3[e];
            }
            if (k + 32 < K) {   // prefetch next K-tile of A (global_prefetch_b8)
                __builtin_prefetch(A + aoff0 + kb + 32, 0, 1);
                __builtin_prefetch(A + aoff1 + kb + 32, 0, 1);
            }
        } else {
            const bf16* A = (const bf16*)Ap;
            v8bf l0 = *(const v8bf*)(A + aoff0 + kb);
            v8bf h0 = *(const v8bf*)(A + aoff0 + kb + 16);
            v8bf l1 = *(const v8bf*)(A + aoff1 + kb);
            v8bf h1 = *(const v8bf*)(A + aoff1 + kb + 16);
#pragma unroll
            for (int e = 0; e < 8; ++e) {
                a0[e] = l0[e]; a0[e + 8] = h0[e];
                a1[e] = l1[e]; a1[e + 8] = h1[e];
            }
            if (k + 32 < K) {
                __builtin_prefetch(A + aoff0 + kb + 32, 0, 1);
                __builtin_prefetch(A + aoff1 + kb + 32, 0, 1);
            }
        }
#pragma unroll
        for (int j = 0; j < 4; ++j) {
            const bf16* wp = W + (long)(n0 + (j << 4) + r) * ldw + kb;
            v8bf lo = *(const v8bf*)(wp);
            v8bf hi = *(const v8bf*)(wp + 16);
            if (k + 32 < K) __builtin_prefetch(wp + 32, 0, 1);
            v16bf bfrag;
#pragma unroll
            for (int e = 0; e < 8; ++e) { bfrag[e] = lo[e]; bfrag[e + 8] = hi[e]; }
            acc[0][j] = __builtin_amdgcn_wmma_f32_16x16x32_bf16(
                false, a0, false, bfrag, (short)0, acc[0][j], false, false);
            acc[1][j] = __builtin_amdgcn_wmma_f32_16x16x32_bf16(
                false, a1, false, bfrag, (short)0, acc[1][j], false, false);
        }
    }

#pragma unroll
    for (int ti = 0; ti < 2; ++ti) {
#pragma unroll
        for (int j = 0; j < 4; ++j) {
            int col = n0 + (j << 4) + r;
            float bv = bias ? bias[col] : 0.0f;
#pragma unroll
            for (int i = 0; i < 8; ++i) {
                int row = m0 + (ti << 4) + i + (g << 3);
                float v = acc[ti][j][i] + bv;
                long idx = (long)row * ldc + col;
                if constexpr (MODE == 0) {
                    ((float*)Cp)[idx] = v;
                } else if constexpr (MODE == 1) {
                    ((bf16*)Cp)[idx] = (bf16)v;
                } else if constexpr (MODE == 2) {
                    ((bf16*)Cp)[idx] = (bf16)gelu_f(v);
                } else {
                    ((float*)Cp)[idx] = v + res[idx];
                }
            }
        }
    }
}

// ------------------------------------------------------------------
// per-block std (ddof=1) and max over 64 tokens x 1536 features -> bf16
// ------------------------------------------------------------------
__global__ void k_blockstats(const float* __restrict__ xf,
                             bf16* __restrict__ sd, bf16* __restrict__ mx)
{
    int blk = blockIdx.x;                               // b*128 + block (1024)
    int f = blockIdx.y * 256 + threadIdx.x;             // 0..1535
    long base = (long)blk * 64 * 1536;
    float s = 0.f, ss = 0.f, m = -1e30f;
    for (int j = 0; j < 64; ++j) {
        float v = xf[base + (long)j * 1536 + f];
        s += v; ss += v * v; m = fmaxf(m, v);
    }
    float mean = s * (1.0f / 64.0f);
    float var = (ss - 64.0f * mean * mean) * (1.0f / 63.0f);
    if (var < 0.f) var = 0.f;
    sd[(long)blk * 1536 + f] = (bf16)sqrtf(var);
    mx[(long)blk * 1536 + f] = (bf16)m;
}

// ------------------------------------------------------------------
// LayerNorm over D=128, one block per row
// ------------------------------------------------------------------
template<bool BF>
__global__ void k_ln(const float* __restrict__ x, const float* __restrict__ g,
                     const float* __restrict__ b, void* __restrict__ out)
{
    long row = blockIdx.x;
    int d = threadIdx.x;
    float v = x[row * 128 + d];
    __shared__ float sh[128];
    sh[d] = v; __syncthreads();
    for (int o = 64; o > 0; o >>= 1) { if (d < o) sh[d] += sh[d + o]; __syncthreads(); }
    float mean = sh[0] * (1.0f / 128.0f); __syncthreads();
    float c = v - mean;
    sh[d] = c * c; __syncthreads();
    for (int o = 64; o > 0; o >>= 1) { if (d < o) sh[d] += sh[d + o]; __syncthreads(); }
    float var = sh[0] * (1.0f / 128.0f);
    float y = c * rsqrtf(var + 1e-5f) * g[d] + b[d];
    if (BF) ((bf16*)out)[row * 128 + d] = (bf16)y;
    else    ((float*)out)[row * 128 + d] = y;
}

// LN + pos_emb -> pm
__global__ void k_ln_pos(const float* __restrict__ x, const float* __restrict__ g,
                         const float* __restrict__ b, const float* __restrict__ pos,
                         float* __restrict__ pm)
{
    long row = blockIdx.x;          // b*128 + m
    int m = (int)(row & 127);
    int d = threadIdx.x;
    float v = x[row * 128 + d];
    __shared__ float sh[128];
    sh[d] = v; __syncthreads();
    for (int o = 64; o > 0; o >>= 1) { if (d < o) sh[d] += sh[d + o]; __syncthreads(); }
    float mean = sh[0] * (1.0f / 128.0f); __syncthreads();
    float c = v - mean;
    sh[d] = c * c; __syncthreads();
    for (int o = 64; o > 0; o >>= 1) { if (d < o) sh[d] += sh[d + o]; __syncthreads(); }
    float var = sh[0] * (1.0f / 128.0f);
    pm[row * 128 + d] = c * rsqrtf(var + 1e-5f) * g[d] + b[d] + pos[(long)m * 128 + d];
}

__global__ void k_mean_pm(const float* __restrict__ pm, float* __restrict__ st)
{
    int b = blockIdx.x, d = threadIdx.x;
    float a = 0.f;
    for (int m = 0; m < 128; ++m) a += pm[((long)b * 128 + m) * 128 + d];
    st[b * 128 + d] = a * (1.0f / 128.0f);
}

// y[b,d] = dot(x[b,:], w[d,:]) + bias[d]   (8 x 128)
__global__ void k_vecmat(const float* __restrict__ x, const float* __restrict__ w,
                         const float* __restrict__ bi, float* __restrict__ y)
{
    int b = blockIdx.x, d = threadIdx.x;
    __shared__ float xs[128];
    xs[d] = x[b * 128 + d]; __syncthreads();
    float a = 0.f;
    for (int k = 0; k < 128; ++k) a += xs[k] * w[d * 128 + k];
    y[b * 128 + d] = a + bi[d];
}

// scores[b,m] = dot(q[b,:], pm[b,m,:] @ kp^T + kp_b) / sqrt(128)
__global__ void k_score(const float* __restrict__ pm, const float* __restrict__ kpw,
                        const float* __restrict__ kpb, const float* __restrict__ q,
                        float* __restrict__ sc)
{
    int bm = blockIdx.x;           // 0..1023
    int b = bm >> 7;
    int d = threadIdx.x;
    __shared__ float xs[128];
    __shared__ float sh[128];
    xs[d] = pm[(long)bm * 128 + d]; __syncthreads();
    float kd = 0.f;
    for (int k = 0; k < 128; ++k) kd += xs[k] * kpw[d * 128 + k];
    kd += kpb[d];
    sh[d] = kd * q[b * 128 + d]; __syncthreads();
    for (int o = 64; o > 0; o >>= 1) { if (d < o) sh[d] += sh[d + o]; __syncthreads(); }
    if (d == 0) sc[bm] = sh[0] * 0.08838834764831845f;   // 1/sqrt(128)
}

// serial top-16 + softmax(tv/TEMP); one thread per batch (128 candidates)
__global__ void k_topk(const float* __restrict__ sc, float* __restrict__ tw, int* __restrict__ ti)
{
    int b = blockIdx.x;
    float vals[128];
    for (int i = 0; i < 128; ++i) vals[i] = sc[b * 128 + i];
    float tv[16]; int tx[16];
    for (int k = 0; k < 16; ++k) {
        int bi = 0; float bv = vals[0];
        for (int i = 1; i < 128; ++i) if (vals[i] > bv) { bv = vals[i]; bi = i; }
        tv[k] = bv; tx[k] = bi; vals[bi] = -1e30f;
    }
    float m = tv[0] * 0.2f;        // tv sorted desc, TEMP=5
    float se = 0.f, ex[16];
    for (int k = 0; k < 16; ++k) { ex[k] = __expf(tv[k] * 0.2f - m); se += ex[k]; }
    float inv = 1.0f / se;
    for (int k = 0; k < 16; ++k) { tw[b * 16 + k] = ex[k] * inv; ti[b * 16 + k] = tx[k]; }
}

// ctx[(b*16+kk), i, :] : i==0 -> st ; 1..t*64 -> ac ; else extracted window from h
__global__ void k_build_ctx(const float* __restrict__ h, const float* __restrict__ st,
                            const float* __restrict__ ac, const int* __restrict__ ti,
                            float* __restrict__ ctx, int t, int L)
{
    int s = blockIdx.x;            // 0..127 = b*16+kk
    int b = s >> 4, kk = s & 15;
    int i = blockIdx.y;            // 0..L-1
    int d = threadIdx.x;
    float val;
    int nac = t * 64;
    if (i == 0) {
        val = st[b * 128 + d];
    } else if (i <= nac) {
        val = ac[((long)b * 256 + (i - 1)) * 128 + d];
    } else {
        int j = i - 1 - nac;
        int start = ti[b * 16 + kk] * 64 - 32;
        if (start < 0) start = 0;
        if (start > 8192 - 64) start = 8192 - 64;
        val = h[((long)b * 8192 + start + j) * 128 + d];
    }
    ctx[((long)s * L + i) * 128 + d] = val;
}

// append top-1 window to ac
__global__ void k_append_ac(const float* __restrict__ h, const int* __restrict__ ti,
                            float* __restrict__ ac, int t)
{
    int b = blockIdx.x, j = blockIdx.y, d = threadIdx.x;
    int start = ti[b * 16] * 64 - 32;
    if (start < 0) start = 0;
    if (start > 8192 - 64) start = 8192 - 64;
    ac[((long)b * 256 + t * 64 + j) * 128 + d] = h[((long)b * 8192 + start + j) * 128 + d];
}

// foveal self-attention: qkv f32 [(128*L) x 384], heads of 32; out bf16 [(128*L) x 128]
__global__ void k_attn(const float* __restrict__ qkv, bf16* __restrict__ ao, int L)
{
    int seq = blockIdx.x, qi = blockIdx.y, h = blockIdx.z;
    int tid = threadIdx.x;  // 64
    long base = (long)seq * L;
    __shared__ float sh[64];
    __shared__ float qs[32];
    if (tid < 32) qs[tid] = qkv[(base + qi) * 384 + h * 32 + tid];
    __syncthreads();
    const float scale = 0.17677669529663687f;   // 1/sqrt(32)
    float mx = -1e30f;
    for (int j = tid; j < L; j += 64) {
        const float* kp = qkv + (base + j) * 384 + 128 + h * 32;
        float s = 0.f;
        for (int d = 0; d < 32; ++d) s += qs[d] * kp[d];
        mx = fmaxf(mx, s * scale);
    }
    sh[tid] = mx; __syncthreads();
    for (int o = 32; o > 0; o >>= 1) { if (tid < o) sh[tid] = fmaxf(sh[tid], sh[tid + o]); __syncthreads(); }
    mx = sh[0]; __syncthreads();
    float acc[32];
#pragma unroll
    for (int d = 0; d < 32; ++d) acc[d] = 0.f;
    float se = 0.f;
    for (int j = tid; j < L; j += 64) {
        const float* kp = qkv + (base + j) * 384 + 128 + h * 32;
        float s = 0.f;
        for (int d = 0; d < 32; ++d) s += qs[d] * kp[d];
        float p = __expf(s * scale - mx);
        se += p;
        const float* vp = qkv + (base + j) * 384 + 256 + h * 32;
        for (int d = 0; d < 32; ++d) acc[d] += p * vp[d];
    }
    sh[tid] = se; __syncthreads();
    for (int o = 32; o > 0; o >>= 1) { if (tid < o) sh[tid] += sh[tid + o]; __syncthreads(); }
    float inv = 1.0f / sh[0]; __syncthreads();
    for (int d = 0; d < 32; ++d) {
        sh[tid] = acc[d]; __syncthreads();
        for (int o = 32; o > 0; o >>= 1) { if (tid < o) sh[tid] += sh[tid + o]; __syncthreads(); }
        if (tid == 0) ao[(base + qi) * 128 + h * 32 + d] = (bf16)(sh[0] * inv);
        __syncthreads();
    }
}

// cross attention: q [8x128x128], k/v rows b*256+j (j < Lkv); out bf16 [1024 x 128]
__global__ void k_xattn(const float* __restrict__ q, const float* __restrict__ kk,
                        const float* __restrict__ vv, bf16* __restrict__ ao, int Lkv)
{
    int b = blockIdx.x, qi = blockIdx.y, h = blockIdx.z;
    int tid = threadIdx.x;  // 64
    __shared__ float sh[64];
    __shared__ float qs[32];
    if (tid < 32) qs[tid] = q[((long)b * 128 + qi) * 128 + h * 32 + tid];
    __syncthreads();
    const float scale = 0.17677669529663687f;
    long kbase = (long)b * 256;
    float mx = -1e30f;
    for (int j = tid; j < Lkv; j += 64) {
        const float* kp = kk + (kbase + j) * 128 + h * 32;
        float s = 0.f;
        for (int d = 0; d < 32; ++d) s += qs[d] * kp[d];
        mx = fmaxf(mx, s * scale);
    }
    sh[tid] = mx; __syncthreads();
    for (int o = 32; o > 0; o >>= 1) { if (tid < o) sh[tid] = fmaxf(sh[tid], sh[tid + o]); __syncthreads(); }
    mx = sh[0]; __syncthreads();
    float acc[32];
#pragma unroll
    for (int d = 0; d < 32; ++d) acc[d] = 0.f;
    float se = 0.f;
    for (int j = tid; j < Lkv; j += 64) {
        const float* kp = kk + (kbase + j) * 128 + h * 32;
        float s = 0.f;
        for (int d = 0; d < 32; ++d) s += qs[d] * kp[d];
        float p = __expf(s * scale - mx);
        se += p;
        const float* vp = vv + (kbase + j) * 128 + h * 32;
        for (int d = 0; d < 32; ++d) acc[d] += p * vp[d];
    }
    sh[tid] = se; __syncthreads();
    for (int o = 32; o > 0; o >>= 1) { if (tid < o) sh[tid] += sh[tid + o]; __syncthreads(); }
    float inv = 1.0f / sh[0]; __syncthreads();
    for (int d = 0; d < 32; ++d) {
        sh[tid] = acc[d]; __syncthreads();
        for (int o = 32; o > 0; o >>= 1) { if (tid < o) sh[tid] += sh[tid + o]; __syncthreads(); }
        if (tid == 0) ao[((long)b * 128 + qi) * 128 + h * 32 + d] = (bf16)(sh[0] * inv);
        __syncthreads();
    }
}

__global__ void k_copy_cl(const float* __restrict__ ctx, float* __restrict__ cl, int L)
{
    int s = blockIdx.x;
    cl[(long)s * 128 + threadIdx.x] = ctx[(long)s * L * 128 + threadIdx.x];
}

// cl MLP (128 rows): ff1 (128->512) + gelu
__global__ void k_ff1(const float* __restrict__ x, const float* __restrict__ w,
                      const float* __restrict__ bi, float* __restrict__ y)
{
    int row = blockIdx.x;
    int o = threadIdx.x;  // 512
    __shared__ float xs[128];
    if (o < 128) xs[o] = x[(long)row * 128 + o];
    __syncthreads();
    float a = 0.f;
    for (int k = 0; k < 128; ++k) a += xs[k] * w[o * 128 + k];
    y[(long)row * 512 + o] = gelu_f(a + bi[o]);
}

// ff2 (512->128) + residual into cl
__global__ void k_ff2_res(const float* __restrict__ y, const float* __restrict__ w,
                          const float* __restrict__ bi, float* __restrict__ cl)
{
    int row = blockIdx.x;
    int o = threadIdx.x;  // 128
    __shared__ float ys[512];
    for (int k = o; k < 512; k += 128) ys[k] = y[(long)row * 512 + k];
    __syncthreads();
    float a = 0.f;
    for (int k = 0; k < 512; ++k) a += ys[k] * w[o * 512 + k];
    cl[(long)row * 128 + o] += a + bi[o];
}

__global__ void k_st_update(const float* __restrict__ tw, const float* __restrict__ cl,
                            float* __restrict__ st)
{
    int b = blockIdx.x, d = threadIdx.x;
    float a = 0.f;
    for (int k = 0; k < 16; ++k) a += tw[b * 16 + k] * cl[((long)(b * 16 + k)) * 128 + d];
    st[b * 128 + d] = a;
}

__global__ void k_gate(const float* __restrict__ mg1w, const float* __restrict__ mg1b,
                       const float* __restrict__ mg2w, const float* __restrict__ mg2b,
                       float tin, float* __restrict__ gate)
{
    int d = threadIdx.x;  // 128
    __shared__ float sh[128];
    float h = gelu_f(mg1w[d] * tin + mg1b[d]);
    sh[d] = h * mg2w[d];
    __syncthreads();
    for (int o = 64; o > 0; o >>= 1) { if (d < o) sh[d] += sh[d + o]; __syncthreads(); }
    if (d == 0) gate[0] = 1.0f / (1.0f + __expf(-(sh[0] + mg2b[0])));
}

__global__ void k_pm_update(float* __restrict__ pm, const float* __restrict__ d,
                            const float* __restrict__ gate)
{
    long i = (long)blockIdx.x * blockDim.x + threadIdx.x;
    if (i < 131072) pm[i] += gate[0] * d[i];
}

// stage5: v[b,:] = LN(st[b]; on) @ op_w^T + op_b
__global__ void k_out_vec(const float* __restrict__ st, const float* __restrict__ g,
                          const float* __restrict__ b, const float* __restrict__ opw,
                          const float* __restrict__ opb, float* __restrict__ vout)
{
    int bq = blockIdx.x;
    int d = threadIdx.x;
    __shared__ float sh[128];
    __shared__ float lns[128];
    float x = st[bq * 128 + d];
    sh[d] = x; __syncthreads();
    for (int o = 64; o > 0; o >>= 1) { if (d < o) sh[d] += sh[d + o]; __syncthreads(); }
    float mean = sh[0] * (1.0f / 128.0f); __syncthreads();
    float c = x - mean;
    sh[d] = c * c; __syncthreads();
    for (int o = 64; o > 0; o >>= 1) { if (d < o) sh[d] += sh[d + o]; __syncthreads(); }
    float var = sh[0] * (1.0f / 128.0f);
    lns[d] = c * rsqrtf(var + 1e-5f) * g[d] + b[d];
    __syncthreads();
    float a = 0.f;
    for (int k = 0; k < 128; ++k) a += lns[k] * opw[d * 128 + k];
    vout[bq * 128 + d] = a + opb[d];
}

__global__ void k_add_bcast(const float* __restrict__ xs, const float* __restrict__ vvec,
                            float* __restrict__ out)
{
    long i = (long)blockIdx.x * blockDim.x + threadIdx.x;
    if (i >= 8L * 8192 * 128) return;
    int b = (int)(i / (8192L * 128));
    int d = (int)(i & 127);
    out[i] = xs[i] + vvec[b * 128 + d];
}

// ==================================================================
// host orchestration
// ==================================================================
static void gemm_launch(int mode, bool af32, const void* A, long lda, const bf16* W, long ldw,
                        const float* bias, void* C, long ldc, const float* res,
                        int M, int N, int K, hipStream_t s)
{
    int waves = (M / 32) * (N / 64);
    int blocks = (waves + 7) / 8;
    dim3 blk(256);
    if (af32) {
        k_gemm<1, true><<<blocks, blk, 0, s>>>(A, lda, W, ldw, bias, C, ldc, res, M, N, K);
        return;
    }
    switch (mode) {
    case 0: k_gemm<0, false><<<blocks, blk, 0, s>>>(A, lda, W, ldw, bias, C, ldc, res, M, N, K); break;
    case 1: k_gemm<1, false><<<blocks, blk, 0, s>>>(A, lda, W, ldw, bias, C, ldc, res, M, N, K); break;
    case 2: k_gemm<2, false><<<blocks, blk, 0, s>>>(A, lda, W, ldw, bias, C, ldc, res, M, N, K); break;
    default: k_gemm<3, false><<<blocks, blk, 0, s>>>(A, lda, W, ldw, bias, C, ldc, res, M, N, K); break;
    }
}

extern "C" void kernel_launch(void* const* d_in, const int* in_sizes, int n_in,
                              void* d_out, int out_size, void* d_ws, size_t ws_size,
                              hipStream_t stream)
{
    (void)in_sizes; (void)n_in; (void)out_size; (void)ws_size;

    const float* x_sacc = (const float*)d_in[0];   // (8,8192,128)
    const float* x_full = (const float*)d_in[1];   // (8,8192,1536)
    const float* pe_conv_proj_w = (const float*)d_in[2];
    const float* pe_conv_proj_b = (const float*)d_in[3];
    const float* pe_conv_w      = (const float*)d_in[4];
    const float* pe_conv_b      = (const float*)d_in[5];
    const float* pe_std_proj_w  = (const float*)d_in[6];
    const float* pe_std_proj_b  = (const float*)d_in[7];
    const float* pe_max_proj_w  = (const float*)d_in[8];
    const float* pe_max_proj_b  = (const float*)d_in[9];
    const float* pe_project_w   = (const float*)d_in[10];
    const float* pe_project_b   = (const float*)d_in[11];
    const float* pe_norm_g      = (const float*)d_in[12];
    const float* pe_norm_b      = (const float*)d_in[13];
    const float* pe_pos_emb     = (const float*)d_in[14];
    const float* fov_attn_in_w  = (const float*)d_in[15];
    const float* fov_attn_in_b  = (const float*)d_in[16];
    const float* fov_attn_out_w = (const float*)d_in[17];
    const float* fov_attn_out_b = (const float*)d_in[18];
    const float* fov_n1_g       = (const float*)d_in[19];
    const float* fov_n1_b       = (const float*)d_in[20];
    const float* fov_n2_g       = (const float*)d_in[21];
    const float* fov_n2_b       = (const float*)d_in[22];
    const float* fov_ff1_w      = (const float*)d_in[23];
    const float* fov_ff1_b      = (const float*)d_in[24];
    const float* fov_ff2_w      = (const float*)d_in[25];
    const float* fov_ff2_b      = (const float*)d_in[26];
    const float* qp_w  = (const float*)d_in[27];
    const float* qp_b  = (const float*)d_in[28];
    const float* kp_w  = (const float*)d_in[29];
    const float* kp_b  = (const float*)d_in[30];
    const float* ma_in_w  = (const float*)d_in[31];
    const float* ma_in_b  = (const float*)d_in[32];
    const float* ma_out_w = (const float*)d_in[33];
    const float* ma_out_b = (const float*)d_in[34];
    const float* mn_g  = (const float*)d_in[35];
    const float* mn_b  = (const float*)d_in[36];
    const float* mg1_w = (const float*)d_in[37];
    const float* mg1_b = (const float*)d_in[38];
    const float* mg2_w = (const float*)d_in[39];
    const float* mg2_b = (const float*)d_in[40];
    const float* op_w  = (const float*)d_in[41];
    const float* op_b  = (const float*)d_in[42];
    const float* on_g  = (const float*)d_in[43];
    const float* on_b  = (const float*)d_in[44];
    const float* ln1_g = (const float*)d_in[45];
    const float* ln1_b = (const float*)d_in[46];
    const float* ln2_g = (const float*)d_in[47];
    const float* ln2_b = (const float*)d_in[48];
    const float* mlp1_w = (const float*)d_in[49];
    const float* mlp1_b = (const float*)d_in[50];
    const float* mlp2_w = (const float*)d_in[51];
    const float* mlp2_b = (const float*)d_in[52];

    // ---------------- workspace carve-up ----------------
    char* base = (char*)d_ws;
    size_t off = 0;
    auto alloc = [&](size_t bytes) { size_t o = off; off = (off + bytes + 255) & ~(size_t)255; return o; };

    size_t o_w_conv_proj = alloc(256L * 1536 * 2);
    size_t o_w_convp     = alloc(256L * 16384 * 2);
    size_t o_w_stdp      = alloc(256L * 1536 * 2);
    size_t o_w_maxp      = alloc(256L * 1536 * 2);
    size_t o_w_project   = alloc(128L * 768 * 2);
    size_t o_w_attn_in   = alloc(384L * 128 * 2);
    size_t o_w_attn_out  = alloc(128L * 128 * 2);
    size_t o_w_ma_in     = alloc(384L * 128 * 2);
    size_t o_w_ma_out    = alloc(128L * 128 * 2);
    size_t o_w_mlp1      = alloc(512L * 128 * 2);
    size_t o_w_mlp2      = alloc(128L * 512 * 2);
    size_t o_pm    = alloc(1024L * 128 * 4);
    size_t o_st    = alloc(8L * 128 * 4);
    size_t o_h     = alloc(65536L * 128 * 4);
    size_t o_ac    = alloc(8L * 256 * 128 * 4);
    size_t o_ac_bf = alloc(8L * 256 * 128 * 2);
    size_t o_qv    = alloc(8L * 128 * 4);
    size_t o_sc    = alloc(1024L * 4);
    size_t o_tw    = alloc(8L * 16 * 4);
    size_t o_ti    = alloc(8L * 16 * 4);
    size_t o_cl    = alloc(128L * 128 * 4);
    size_t o_clln  = alloc(128L * 128 * 4);
    size_t o_ff1o  = alloc(128L * 512 * 4);
    size_t o_pmn   = alloc(1024L * 128 * 2);
    size_t o_qm    = alloc(1024L * 128 * 4);
    size_t o_kkb   = alloc(2048L * 128 * 4);
    size_t o_vvb   = alloc(2048L * 128 * 4);
    size_t o_ao2   = alloc(1024L * 128 * 2);
    size_t o_dbuf  = alloc(1024L * 128 * 4);
    size_t o_gate  = alloc(256);
    size_t o_vvec  = alloc(8L * 128 * 4);
    size_t o_proj  = alloc(1024L * 128 * 4);
    size_t o_scr   = alloc(90L * 1024 * 1024);   // phase-overlaid scratch

    bf16* W_conv_proj = (bf16*)(base + o_w_conv_proj);
    bf16* W_convp     = (bf16*)(base + o_w_convp);
    bf16* W_stdp      = (bf16*)(base + o_w_stdp);
    bf16* W_maxp      = (bf16*)(base + o_w_maxp);
    bf16* W_project   = (bf16*)(base + o_w_project);
    bf16* W_attn_in   = (bf16*)(base + o_w_attn_in);
    bf16* W_attn_out  = (bf16*)(base + o_w_attn_out);
    bf16* W_ma_in     = (bf16*)(base + o_w_ma_in);
    bf16* W_ma_out    = (bf16*)(base + o_w_ma_out);
    bf16* W_mlp1      = (bf16*)(base + o_w_mlp1);
    bf16* W_mlp2      = (bf16*)(base + o_w_mlp2);
    float* pm   = (float*)(base + o_pm);
    float* st   = (float*)(base + o_st);
    float* h    = (float*)(base + o_h);
    float* ac   = (float*)(base + o_ac);
    bf16*  acbf = (bf16*)(base + o_ac_bf);
    float* qv   = (float*)(base + o_qv);
    float* sc   = (float*)(base + o_sc);
    float* tw   = (float*)(base + o_tw);
    int*   ti   = (int*)(base + o_ti);
    float* cl   = (float*)(base + o_cl);
    float* clln = (float*)(base + o_clln);
    float* ff1o = (float*)(base + o_ff1o);
    bf16*  pmn  = (bf16*)(base + o_pmn);
    float* qm   = (float*)(base + o_qm);
    float* kkb  = (float*)(base + o_kkb);
    float* vvb  = (float*)(base + o_vvb);
    bf16*  ao2  = (bf16*)(base + o_ao2);
    float* dbuf = (float*)(base + o_dbuf);
    float* gate = (float*)(base + o_gate);
    float* vvec = (float*)(base + o_vvec);
    float* proj = (float*)(base + o_proj);
    char*  scr  = base + o_scr;

    // phase A scratch layout
    bf16* xm   = (bf16*)(scr);                               // 65536x256
    bf16* sdb  = (bf16*)(scr + 33554432);                    // 1024x1536
    bf16* mxb  = (bf16*)(scr + 33554432 + 3145728);          // 1024x1536
    bf16* comb = (bf16*)(scr + 33554432 + 2 * 3145728);      // 1024x768
    // loop scratch layout (sized for Lmax=257)
    float* ctx  = (float*)(scr);                             // 128*257*128 f32
    bf16*  nrm  = (bf16*)(scr + 16842752);                   // 128*257*128 bf16
    float* qkv  = (float*)(scr + 16842752 + 8421376);        // 128*257*384 f32
    bf16*  ao   = (bf16*)(scr + 16842752 + 8421376 + 50528256);
    // final scratch layout
    bf16* lnx    = (bf16*)(scr);                             // 65536x128
    bf16* hidden = (bf16*)(scr + 16777216);                  // 65536x512

    auto cvt = [&](const float* s, bf16* d, long n) {
        int blocks = (int)((n + 255) / 256);
        k_cvt_bf16<<<blocks, 256, 0, stream>>>(s, d, n);
    };

    // ------------- weight conversion (bf16) -------------
    cvt(pe_conv_proj_w, W_conv_proj, 256L * 1536);
    k_permute_convw<<<16384, 256, 0, stream>>>(pe_conv_w, W_convp);
    cvt(pe_std_proj_w, W_stdp, 256L * 1536);
    cvt(pe_max_proj_w, W_maxp, 256L * 1536);
    cvt(pe_project_w, W_project, 128L * 768);
    cvt(fov_attn_in_w, W_attn_in, 384L * 128);
    cvt(fov_attn_out_w, W_attn_out, 128L * 128);
    cvt(ma_in_w, W_ma_in, 384L * 128);
    cvt(ma_out_w, W_ma_out, 128L * 128);
    cvt(mlp1_w, W_mlp1, 512L * 128);
    cvt(mlp2_w, W_mlp2, 128L * 512);

    // h = LN(x_sacc; ln1)
    k_ln<false><<<65536, 128, 0, stream>>>(x_sacc, ln1_g, ln1_b, h);

    // ------------- peripheral -------------
    // xm = x_full @ conv_proj_w^T + b  (bf16 out)   M=65536 K=1536 N=256
    gemm_launch(1, true, x_full, 1536, W_conv_proj, 1536, pe_conv_proj_b,
                xm, 256, nullptr, 65536, 256, 1536, stream);
    k_blockstats<<<dim3(1024, 6), 256, 0, stream>>>(x_full, sdb, mxb);
    // conv: M=1024 K=16384 N=256 -> comb cols [0,256)
    gemm_launch(1, false, xm, 16384, W_convp, 16384, pe_conv_b,
                comb, 768, nullptr, 1024, 256, 16384, stream);
    // std_proj -> comb cols [256,512)
    gemm_launch(1, false, sdb, 1536, W_stdp, 1536, pe_std_proj_b,
                comb + 256, 768, nullptr, 1024, 256, 1536, stream);
    // max_proj -> comb cols [512,768)
    gemm_launch(1, false, mxb, 1536, W_maxp, 1536, pe_max_proj_b,
                comb + 512, 768, nullptr, 1024, 256, 1536, stream);
    // project: M=1024 K=768 N=128 -> f32
    gemm_launch(0, false, comb, 768, W_project, 768, pe_project_b,
                proj, 128, nullptr, 1024, 128, 768, stream);
    k_ln_pos<<<1024, 128, 0, stream>>>(proj, pe_norm_g, pe_norm_b, pe_pos_emb, pm);
    k_mean_pm<<<8, 128, 0, stream>>>(pm, st);

    // ------------- saccade loop -------------
    for (int t = 0; t < 4; ++t) {
        int L = 65 + t * 64;          // 1 + t*64 + 64
        int rows = 128 * L;

        k_vecmat<<<8, 128, 0, stream>>>(st, qp_w, qp_b, qv);
        k_score<<<1024, 128, 0, stream>>>(pm, kp_w, kp_b, qv, sc);
        k_topk<<<8, 1, 0, stream>>>(sc, tw, ti);

        k_build_ctx<<<dim3(128, L), 128, 0, stream>>>(h, st, ac, ti, ctx, t, L);
        k_append_ac<<<dim3(8, 64), 128, 0, stream>>>(h, ti, ac, t);

        k_ln<true><<<rows, 128, 0, stream>>>(ctx, fov_n1_g, fov_n1_b, nrm);
        gemm_launch(0, false, nrm, 128, W_attn_in, 128, fov_attn_in_b,
                    qkv, 384, nullptr, rows, 384, 128, stream);
        k_attn<<<dim3(128, L, 4), 64, 0, stream>>>(qkv, ao, L);
        gemm_launch(3, false, ao, 128, W_attn_out, 128, fov_attn_out_b,
                    ctx, 128, ctx, rows, 128, 128, stream);

        k_copy_cl<<<128, 128, 0, stream>>>(ctx, cl, L);
        k_ln<false><<<128, 128, 0, stream>>>(cl, fov_n2_g, fov_n2_b, clln);
        k_ff1<<<128, 512, 0, stream>>>(clln, fov_ff1_w, fov_ff1_b, ff1o);
        k_ff2_res<<<128, 128, 0, stream>>>(ff1o, fov_ff2_w, fov_ff2_b, cl);
        k_st_update<<<8, 128, 0, stream>>>(tw, cl, st);

        // cross attention: d = MHA(LN(pm; mn), ac, ac)
        k_ln<true><<<1024, 128, 0, stream>>>(pm, mn_g, mn_b, pmn);
        cvt(ac, acbf, 8L * 256 * 128);
        gemm_launch(0, false, pmn, 128, W_ma_in, 128, ma_in_b,
                    qm, 128, nullptr, 1024, 128, 128, stream);
        gemm_launch(0, false, acbf, 128, W_ma_in + 128L * 128, 128, ma_in_b + 128,
                    kkb, 128, nullptr, 2048, 128, 128, stream);
        gemm_launch(0, false, acbf, 128, W_ma_in + 256L * 128, 128, ma_in_b + 256,
                    vvb, 128, nullptr, 2048, 128, 128, stream);
        k_xattn<<<dim3(8, 128, 4), 64, 0, stream>>>(qm, kkb, vvb, ao2, (t + 1) * 64);
        gemm_launch(0, false, ao2, 128, W_ma_out, 128, ma_out_b,
                    dbuf, 128, nullptr, 1024, 128, 128, stream);

        k_gate<<<1, 128, 0, stream>>>(mg1_w, mg1_b, mg2_w, mg2_b, (float)t * 0.25f, gate);
        k_pm_update<<<512, 256, 0, stream>>>(pm, dbuf, gate);
    }

    // ------------- output head -------------
    float* xout = (float*)d_out;
    k_out_vec<<<8, 128, 0, stream>>>(st, on_g, on_b, op_w, op_b, vvec);
    k_add_bcast<<<32768, 256, 0, stream>>>(x_sacc, vvec, xout);
    k_ln<true><<<65536, 128, 0, stream>>>(xout, ln2_g, ln2_b, lnx);
    gemm_launch(2, false, lnx, 128, W_mlp1, 128, mlp1_b,
                hidden, 512, nullptr, 65536, 512, 128, stream);
    gemm_launch(3, false, hidden, 512, W_mlp2, 512, mlp2_b,
                xout, 128, xout, 65536, 128, 512, stream);
}
